// MultiHeadSelfAttention_63264868270276
// MI455X (gfx1250) — compile-verified, matrix operand order checked by reading, not run on hardware
//
#include <hip/hip_runtime.h>

typedef __attribute__((ext_vector_type(16))) _Float16 v16h;
typedef __attribute__((ext_vector_type(8)))  float    v8f;
typedef _Float16 h16;

constexpr int kE = 1024;   // embed dim
constexpr int kS = 2048;   // seq len
constexpr int kB = 4;      // batch
constexpr int kH = 16;     // heads
constexpr int kD = 64;     // head dim
constexpr int kM = kB * kS; // 8192 rows

// ---------------------------------------------------------------------------
// Async global->LDS copy of 16 bytes (per lane), tracked by ASYNCcnt.
// Emitted via inline asm so the operand semantics are exactly the ISA's:
//   global_load_async_to_lds_b128 vdst(LDS byte offset), vaddr(64b), off
// Per the CDNA5 aperture rules, the low 32 bits of a generic shared-memory
// address are the LDS byte address, so a truncation gives VDST directly.
// ---------------------------------------------------------------------------
__device__ __forceinline__ void copy_b128_g2l(const h16* g, h16* l) {
  unsigned loff = (unsigned)(unsigned long long)l;
  asm volatile("global_load_async_to_lds_b128 %0, %1, off"
               :: "v"(loff), "v"(g) : "memory");
}

__device__ __forceinline__ void wait_async0() {
#if __has_builtin(__builtin_amdgcn_s_wait_asynccnt)
  __builtin_amdgcn_s_wait_asynccnt(0);
#else
  asm volatile("s_wait_asynccnt 0x0" ::: "memory");
#endif
}

// ---------------------------------------------------------------------------
// Fragment loader for 16-bit WMMA operands.
// A-matrix 16x32 (MxK) layout per CDNA5 ISA: lane L holds row M=L%16; per-VGPR
// K pairs: lanes 0-15 V0..3 -> K=0..7, V4..7 -> K=16..23; lanes 16-31 get +8.
// B-matrix 32x16 (KxN) is mirrored with N=L%16 and the same K mapping.
// ---------------------------------------------------------------------------
__device__ __forceinline__ v16h load_frag(const h16* p, int row, int ld,
                                          int koff, int lh) {
  union { v16h v; unsigned u[8]; } r;
  const h16* q = p + row * ld + koff;
#pragma unroll
  for (int i = 0; i < 8; ++i) {
    int kb = (i < 4) ? (8 * lh + 2 * i) : (16 + 8 * lh + 2 * (i - 4));
    r.u[i] = *(const unsigned*)(q + kb);
  }
  return r.v;
}

__device__ __forceinline__ v8f wmma_f16(v16h a, v16h b, v8f c) {
  return __builtin_amdgcn_wmma_f32_16x16x32_f16(false, a, false, b,
                                                (short)0, c, false, false);
}

// ---------------------------------------------------------------------------
// fp32 -> fp16 grid-stride convert
// ---------------------------------------------------------------------------
__global__ void cvt_f32_f16(const float* __restrict__ s, h16* __restrict__ d,
                            int n) {
  int i = blockIdx.x * blockDim.x + threadIdx.x;
  int stride = gridDim.x * blockDim.x;
  for (; i < n; i += stride) d[i] = (h16)s[i];
}

// ---------------------------------------------------------------------------
// Tiled WMMA GEMM: C[m,n] = sum_k A[m,k] * W[n,k]  (i.e. A @ W^T)
// Block tile 128x128x32, 256 threads = 8 waves, wave tile 32(M) x 64(N).
// Double-buffered LDS with async global->LDS fills overlapped with WMMA.
// MODE 0: N=3072 fused QKV -> bias + scatter f16 into [B,H,S,D] q/k/v.
// MODE 1: N=1024 out-proj  -> bias + f32 row-major output.
// ---------------------------------------------------------------------------
template <int MODE>
__global__ __launch_bounds__(256) void gemm_wmma(
    const h16* __restrict__ A, const h16* __restrict__ W,
    const float* __restrict__ b0, const float* __restrict__ b1,
    const float* __restrict__ b2, h16* __restrict__ oq, h16* __restrict__ ok,
    h16* __restrict__ ov, float* __restrict__ outf, int N, int K) {
  constexpr int LDT = 40;  // 32 halfs + 8 pad
  __shared__ h16 As[2][128 * LDT];
  __shared__ h16 Bs[2][128 * LDT];

  const int tid = threadIdx.x;
  const int lane = tid & 31;
  const int wid = tid >> 5;
  const int wm = wid >> 1;  // 0..3, 32 rows each
  const int wn = wid & 1;   // 0..1, 64 cols each
  const int lh = lane >> 4;
  const int ln = lane & 15;
  const int m0 = blockIdx.y * 128;
  const int n0 = blockIdx.x * 128;

  const int lrow = tid >> 1;          // 0..127
  const int lseg = (tid & 1) * 16;    // half-row segment (16 halfs)
  const int lofs = lrow * LDT + lseg;

  const h16* gA = A + (size_t)(m0 + lrow) * K + lseg;
  const h16* gB = W + (size_t)(n0 + lrow) * K + lseg;

  v8f acc[2][4];
#pragma unroll
  for (int i = 0; i < 2; ++i)
#pragma unroll
    for (int j = 0; j < 4; ++j) acc[i][j] = {};

  // prologue: stage tile 0 into buffer 0
  copy_b128_g2l(gA, &As[0][lofs]);
  copy_b128_g2l(gA + 8, &As[0][lofs + 8]);
  copy_b128_g2l(gB, &Bs[0][lofs]);
  copy_b128_g2l(gB + 8, &Bs[0][lofs + 8]);

  int cur = 0;
  for (int k0 = 0; k0 < K; k0 += 32) {
    wait_async0();     // this wave's fills of buffer `cur` have landed
    __syncthreads();   // whole tile present; prior reads of other buf retired

    const int nk = k0 + 32;
    if (nk < K) {      // kick off next tile's DMA, overlapped with compute
      const int nb = cur ^ 1;
      copy_b128_g2l(gA + nk, &As[nb][lofs]);
      copy_b128_g2l(gA + nk + 8, &As[nb][lofs + 8]);
      copy_b128_g2l(gB + nk, &Bs[nb][lofs]);
      copy_b128_g2l(gB + nk + 8, &Bs[nb][lofs + 8]);
    }

    v16h af[2], bf[4];
#pragma unroll
    for (int mt = 0; mt < 2; ++mt)
      af[mt] = load_frag(As[cur], wm * 32 + mt * 16 + ln, LDT, 0, lh);
#pragma unroll
    for (int nt = 0; nt < 4; ++nt)
      bf[nt] = load_frag(Bs[cur], wn * 64 + nt * 16 + ln, LDT, 0, lh);
#pragma unroll
    for (int mt = 0; mt < 2; ++mt)
#pragma unroll
      for (int nt = 0; nt < 4; ++nt)
        acc[mt][nt] = wmma_f16(af[mt], bf[nt], acc[mt][nt]);

    cur ^= 1;
  }

  // Epilogue. C layout: element r of v8f = row (r + 8*lh), col = ln.
#pragma unroll
  for (int mt = 0; mt < 2; ++mt)
#pragma unroll
    for (int nt = 0; nt < 4; ++nt)
#pragma unroll
      for (int r = 0; r < 8; ++r) {
        int m = m0 + wm * 32 + mt * 16 + r + 8 * lh;
        int n = n0 + wn * 64 + nt * 16 + ln;
        float v = acc[mt][nt][r];
        if (MODE == 0) {
          int which = n >> 10;
          int e = n & 1023;
          const float* bias = (which == 0) ? b0 : (which == 1) ? b1 : b2;
          v += bias[e];
          h16* dst = (which == 0) ? oq : (which == 1) ? ok : ov;
          int hh = e >> 6, d = e & 63;
          int b = m >> 11, s = m & 2047;
          dst[(((size_t)(b * kH + hh)) * kS + s) * kD + d] = (h16)v;
        } else {
          v += b0[n];
          outf[(size_t)m * N + n] = v;
        }
      }
}

// ---------------------------------------------------------------------------
// Flash-attention: one WG = 128 query rows of one (b,h); 8 waves x 16 rows.
// Loop over 64-key tiles: S = Q K^T (WMMA), online softmax (shfl_xor row
// reductions within 16-lane halves), P staged via LDS into A-fragments,
// O += P V (WMMA). K tile filled with async DMA; V transposed via registers.
// ---------------------------------------------------------------------------
__global__ __launch_bounds__(256) void attn_wmma(
    const h16* __restrict__ Q, const h16* __restrict__ Kh,
    const h16* __restrict__ V, const int* __restrict__ mask,
    h16* __restrict__ ctx) {
  constexpr int LDK = 72;  // 64 halfs + 8 pad
  __shared__ h16 Ks[64 * LDK];      // K tile, [t][d]
  __shared__ h16 Vts[64 * LDK];     // V tile transposed, [d][t]
  __shared__ h16 Ps[8][16 * LDK];   // per-wave P tile, [m][t]

  const int tid = threadIdx.x;
  const int lane = tid & 31;
  const int wid = tid >> 5;
  const int lh = lane >> 4;
  const int ln = lane & 15;
  const int bh = blockIdx.y;  // b*H + h
  const int b = bh / kH, hh = bh % kH;
  const int sblk = blockIdx.x * 128;
  const size_t base = (size_t)bh * kS * kD;

  // Q fragments for this wave's 16 rows (K = 0..31, 32..63)
  const int qrow = sblk + wid * 16 + ln;
  v16h aq0 = load_frag(Q + base, qrow, kD, 0, lh);
  v16h aq1 = load_frag(Q + base, qrow, kD, 32, lh);

  float mrow[8], lrow[8];
  v8f o[4];
#pragma unroll
  for (int r = 0; r < 8; ++r) { mrow[r] = -3.0e38f; lrow[r] = 0.0f; }
#pragma unroll
  for (int dt = 0; dt < 4; ++dt) o[dt] = {};

  const float scale = 0.125f;  // 1/sqrt(64)

  const int trow = tid >> 2;          // 0..63
  const int tseg = (tid & 3) * 16;    // 16-half segment

  for (int t0 = 0; t0 < kS; t0 += 64) {
    __syncthreads();
    {
      // K tile via async DMA
      const h16* gk = Kh + base + (size_t)(t0 + trow) * kD + tseg;
      copy_b128_g2l(gk, &Ks[trow * LDK + tseg]);
      copy_b128_g2l(gk + 8, &Ks[trow * LDK + tseg + 8]);
      // V tile transposed through registers
      const uint4* gv =
          (const uint4*)(V + base + (size_t)(t0 + trow) * kD + tseg);
      union { uint4 q[2]; h16 e[16]; } vv;
      vv.q[0] = gv[0];
      vv.q[1] = gv[1];
#pragma unroll
      for (int i = 0; i < 16; ++i)
        Vts[(tseg + i) * LDK + trow] = vv.e[i];
    }
    wait_async0();
    __syncthreads();

    // scores: 16 rows x 64 keys, per lane element (r+8*lh, nt*16+ln)
    float sc[4][8];
#pragma unroll
    for (int nt = 0; nt < 4; ++nt) {
      v8f c = {};
      v16h bk0 = load_frag(Ks, nt * 16 + ln, LDK, 0, lh);
      v16h bk1 = load_frag(Ks, nt * 16 + ln, LDK, 32, lh);
      c = wmma_f16(aq0, bk0, c);
      c = wmma_f16(aq1, bk1, c);
      int t = t0 + nt * 16 + ln;
      int mk = mask[b * kS + t];
#pragma unroll
      for (int r = 0; r < 8; ++r)
        sc[nt][r] = mk ? c[r] * scale : -3.0e38f;
    }

    // online softmax per row (rows live in 16-lane halves)
    float alpha[8];
#pragma unroll
    for (int r = 0; r < 8; ++r) {
      float mx = sc[0][r];
#pragma unroll
      for (int nt = 1; nt < 4; ++nt) mx = fmaxf(mx, sc[nt][r]);
#pragma unroll
      for (int d = 1; d < 16; d <<= 1)
        mx = fmaxf(mx, __shfl_xor(mx, d, 32));
      float mnew = fmaxf(mrow[r], mx);
      alpha[r] = __expf(mrow[r] - mnew);
      float rs = 0.0f;
#pragma unroll
      for (int nt = 0; nt < 4; ++nt) {
        float p = __expf(sc[nt][r] - mnew);
        sc[nt][r] = p;
        rs += p;
      }
#pragma unroll
      for (int d = 1; d < 16; d <<= 1) rs += __shfl_xor(rs, d, 32);
      lrow[r] = lrow[r] * alpha[r] + rs;
      mrow[r] = mnew;
    }

    // rescale O, stage P into LDS (row-major [m][t]) for A-fragment reload
#pragma unroll
    for (int dt = 0; dt < 4; ++dt)
#pragma unroll
      for (int r = 0; r < 8; ++r) o[dt][r] *= alpha[r];
#pragma unroll
    for (int nt = 0; nt < 4; ++nt)
#pragma unroll
      for (int r = 0; r < 8; ++r)
        Ps[wid][(r + 8 * lh) * LDK + nt * 16 + ln] = (h16)sc[nt][r];
    __syncthreads();

    // O += P @ V
    v16h ap0 = load_frag(Ps[wid], ln, LDK, 0, lh);
    v16h ap1 = load_frag(Ps[wid], ln, LDK, 32, lh);
#pragma unroll
    for (int dt = 0; dt < 4; ++dt) {
      v16h bv0 = load_frag(Vts, dt * 16 + ln, LDK, 0, lh);
      v16h bv1 = load_frag(Vts, dt * 16 + ln, LDK, 32, lh);
      o[dt] = wmma_f16(ap0, bv0, o[dt]);
      o[dt] = wmma_f16(ap1, bv1, o[dt]);
    }
  }

  // normalize + write context [B,S,E] as f16
#pragma unroll
  for (int r = 0; r < 8; ++r) {
    float inv = 1.0f / lrow[r];
    int srow = sblk + wid * 16 + r + 8 * lh;
#pragma unroll
    for (int dt = 0; dt < 4; ++dt) {
      float vo = o[dt][r] * inv;
      ctx[((size_t)(b * kS + srow)) * kE + hh * kD + dt * 16 + ln] = (h16)vo;
    }
  }
}

// ---------------------------------------------------------------------------
extern "C" void kernel_launch(void* const* d_in, const int* in_sizes, int n_in,
                              void* d_out, int out_size, void* d_ws,
                              size_t ws_size, hipStream_t stream) {
  const float* x = (const float*)d_in[0];
  const int* mask = (const int*)d_in[1];
  const float* wq = (const float*)d_in[2];
  const float* bq = (const float*)d_in[3];
  const float* wk = (const float*)d_in[4];
  const float* bk = (const float*)d_in[5];
  const float* wv = (const float*)d_in[6];
  const float* bv = (const float*)d_in[7];
  const float* wo = (const float*)d_in[8];
  const float* bo = (const float*)d_in[9];

  h16* ws = (h16*)d_ws;
  size_t off = 0;
  h16* xh = ws + off;   off += (size_t)kM * kE;        // also reused as ctx
  h16* wqkv = ws + off; off += 3ull * kE * kE;
  h16* woh = ws + off;  off += (size_t)kE * kE;
  h16* qh = ws + off;   off += (size_t)kB * kH * kS * kD;
  h16* kh = ws + off;   off += (size_t)kB * kH * kS * kD;
  h16* vh = ws + off;   off += (size_t)kB * kH * kS * kD;
  h16* ctxh = xh;  // x no longer needed after QKV GEMM

  const int ee = kE * kE;
  cvt_f32_f16<<<4096, 256, 0, stream>>>(x, xh, kM * kE);
  cvt_f32_f16<<<2048, 256, 0, stream>>>(wq, wqkv, ee);
  cvt_f32_f16<<<2048, 256, 0, stream>>>(wk, wqkv + ee, ee);
  cvt_f32_f16<<<2048, 256, 0, stream>>>(wv, wqkv + 2 * ee, ee);
  cvt_f32_f16<<<2048, 256, 0, stream>>>(wo, woh, ee);

  // QKV: M=8192, N=3072, K=1024
  gemm_wmma<0><<<dim3(3 * kE / 128, kM / 128), 256, 0, stream>>>(
      xh, wqkv, bq, bk, bv, qh, kh, vh, nullptr, 3 * kE, kE);

  // attention: grid (S/128, B*H)
  attn_wmma<<<dim3(kS / 128, kB * kH), 256, 0, stream>>>(qh, kh, vh, mask,
                                                         ctxh);

  // out-proj: M=8192, N=1024, K=1024
  gemm_wmma<1><<<dim3(kE / 128, kM / 128), 256, 0, stream>>>(
      ctxh, woh, bo, nullptr, nullptr, nullptr, nullptr, nullptr,
      (float*)d_out, kE, kE);
}
